// LSTM_47923245088969
// MI455X (gfx1250) — matrix operand order repaired and checked by reading, not compiled, not run on hardware
//
#include <hip/hip_runtime.h>

#define BATCH 128
#define TLEN  512
#define HID   256
#define FUT   64
#define TF    (TLEN + FUT)
#define NWG   16
#define TPB   256

typedef __attribute__((ext_vector_type(16))) _Float16 v16h;
typedef __attribute__((ext_vector_type(8)))  float    v8f;

union FragU { v16h v; uint4 q[2]; };

// -------- workspace layout (bytes, all 256-aligned) --------
#define OFF_BAR   0u        // 2 x u32 barrier state
#define OFF_OUTP  256u      // 16*128 f32 readout partials      (8192 B)
#define OFF_B1S   8448u     // 1024 f32  b_ih1+b_hh1            (4096 B)
#define OFF_B2S   12544u    // 1024 f32  b_ih2+b_hh2            (4096 B)
#define OFF_H1    16640u    // f16 [2][128][256] ping-pong      (131072 B)
#define OFF_H2    147712u   // f16 [2][128][256] ping-pong      (131072 B)
#define OFF_WB1   278784u   // packed f16 B-frags layer1        (524288 B)
#define OFF_WB2   803072u   // packed f16 B-frags layer2 K=512  (1048576 B)
// total ~1.81 MB

#define LDS_BYTES ((32 + 64) * 512 * 2)   // 96 KB of 320 KB/WGP

// ---------------- device helpers ----------------
__device__ __forceinline__ float fsig(float x) {
  return 1.0f / (1.0f + __expf(-x));
}
__device__ __forceinline__ float ftanh(float x) {
  x = fminf(fmaxf(x, -15.0f), 15.0f);
  const float e = __expf(2.0f * x);
  return (e - 1.0f) / (e + 1.0f);
}

// A fragment: 16x32 f16 tile of h (row-major [128][256]).
// lane l: row m0+(l&15); halves 0..7 = K kbase+(l>=16?8:0)+0..7,
//         halves 8..15 = K kbase+16+(l>=16?8:0)+0..7  (two 128-bit loads)
__device__ __forceinline__ v16h load_a_frag(const _Float16* __restrict__ h,
                                            int m0, int kbase, int lane) {
  const int hsel = (lane >> 4) & 1;
  const int m    = m0 + (lane & 15);
  const _Float16* p = h + m * HID + kbase + hsel * 8;
  FragU f;
  f.q[0] = *(const uint4*)(p);
  f.q[1] = *(const uint4*)(p + 16);
  return f.v;
}

// B fragment from LDS: frag interior layout [chunk(2)][lane(32)][8 halves]
// -> each ds_load_b128 has 16B lane stride: conflict-free on 64 banks
__device__ __forceinline__ v16h load_b_lds(const _Float16* sm, int frag, int lane) {
  const _Float16* p = sm + frag * 512 + lane * 8;
  FragU f;
  f.q[0] = *(const uint4*)(p);
  f.q[1] = *(const uint4*)(p + 256);
  return f.v;
}

// grid-wide sense/generation barrier (16 blocks on 256 WGPs: always resident)
__device__ __forceinline__ void gbar(unsigned* bar, unsigned gen) {
  __threadfence();
  __syncthreads();
  if (threadIdx.x == 0) {
    const unsigned prev = atomicAdd(&bar[0], 1u);
    if (prev == NWG - 1) {
      atomicExch(&bar[0], 0u);
      __threadfence();
      atomicAdd(&bar[1], 1u);
    } else {
      while (__hip_atomic_load(&bar[1], __ATOMIC_RELAXED,
                               __HIP_MEMORY_SCOPE_AGENT) <= gen) {
        __builtin_amdgcn_s_sleep(1);
      }
    }
    __threadfence();
  }
  __syncthreads();
}

// ---------------- prep kernels ----------------
__global__ void lstm_init(const float* __restrict__ bih1, const float* __restrict__ bhh1,
                          const float* __restrict__ bih2, const float* __restrict__ bhh2,
                          float* __restrict__ b1s, float* __restrict__ b2s,
                          _Float16* __restrict__ h1buf, _Float16* __restrict__ h2buf,
                          unsigned* __restrict__ bar) {
  const int tid = blockIdx.x * blockDim.x + threadIdx.x;
  if (tid < 4 * HID) { b1s[tid] = bih1[tid] + bhh1[tid]; b2s[tid] = bih2[tid] + bhh2[tid]; }
  if (tid < 2 * BATCH * HID) { h1buf[tid] = (_Float16)0.0f; h2buf[tid] = (_Float16)0.0f; }
  if (tid < 2) bar[tid] = 0u;
}

// fragment-interior store index: element (lane, h) -> (h/8)*256 + lane*8 + (h%8)
__device__ __forceinline__ int frag_elem(int lane, int h) {
  return ((h >> 3) << 8) + lane * 8 + (h & 7);
}

// pack W_hh1 [1024][256] -> B-frag order [nT(64)][kT(8)]{[chunk][lane][8]} f16
__global__ void pack_w1(const float* __restrict__ Whh1, _Float16* __restrict__ WB1) {
  const int tid  = blockIdx.x * 256 + threadIdx.x;   // 262144 total
  const int h    = tid & 15;
  const int lane = (tid >> 4) & 31;
  const int kT   = (tid >> 9) & 7;
  const int nT   = tid >> 12;
  const int n = nT * 16 + (lane & 15);
  const int k = kT * 32 + (h >> 3) * 16 + ((lane >> 4) & 1) * 8 + (h & 7);
  WB1[(nT * 8 + kT) * 512 + frag_elem(lane, h)] = (_Float16)Whh1[n * HID + k];
}

// pack [W_ih2; W_hh2] as K=512 -> [nT(64)][kT(16)]{[chunk][lane][8]} f16
__global__ void pack_w2(const float* __restrict__ Wih2, const float* __restrict__ Whh2,
                        _Float16* __restrict__ WB2) {
  const int tid  = blockIdx.x * 256 + threadIdx.x;   // 524288 total
  const int h    = tid & 15;
  const int lane = (tid >> 4) & 31;
  const int kT   = (tid >> 9) & 15;
  const int nT   = tid >> 13;
  const int n = nT * 16 + (lane & 15);
  const int k = kT * 32 + (h >> 3) * 16 + ((lane >> 4) & 1) * 8 + (h & 7);
  const float v = (k < HID) ? Wih2[n * HID + k] : Whh2[n * HID + (k - HID)];
  WB2[(nT * 16 + kT) * 512 + frag_elem(lane, h)] = (_Float16)v;
}

// ---------------- persistent LSTM kernel ----------------
__global__ __launch_bounds__(TPB, 1) void lstm_persistent(
    const float* __restrict__ x_train, const float* __restrict__ wih1,
    const float* __restrict__ wlin,    const float* __restrict__ blin,
    const _Float16* __restrict__ WB1,  const _Float16* __restrict__ WB2,
    const float* __restrict__ b1s,     const float* __restrict__ b2s,
    _Float16* __restrict__ h1buf,     _Float16* __restrict__ h2buf,
    float* __restrict__ outpart,       unsigned* __restrict__ bar,
    float* __restrict__ out) {
  const int lane = threadIdx.x & 31;
  const int wave = threadIdx.x >> 5;   // m-tile 0..7
  const int jT   = blockIdx.x;         // j-tile 0..15
  const int m0   = wave * 16;
  const int j0   = jT * 16;
  const int lj   = lane & 15;
  const int hsel = (lane >> 4) & 1;

  // ---- stage this block's weight fragments into LDS (once) ----
  extern __shared__ _Float16 smem[];
  _Float16* sWB1 = smem;                 // 32 frags * 512 halves (32 KB)
  _Float16* sWB2 = smem + 32 * 512;      // 64 frags * 512 halves (64 KB)
  {
    const uint4* g1 = (const uint4*)WB1;
    const uint4* g2 = (const uint4*)WB2;
    uint4* s1 = (uint4*)sWB1;
    uint4* s2 = (uint4*)sWB2;
    for (int i = threadIdx.x; i < 32 * 64; i += TPB) {     // 32 KB
      const int lf = i >> 6, rem = i & 63;
      const int g = lf >> 3, kb = lf & 7;
      s1[i] = g1[((g * 16 + jT) * 8 + kb) * 64 + rem];
    }
    for (int i = threadIdx.x; i < 64 * 64; i += TPB) {     // 64 KB
      const int lf = i >> 6, rem = i & 63;
      const int g = lf >> 4, kb = lf & 15;
      s2[i] = g2[((g * 16 + jT) * 16 + kb) * 64 + rem];
    }
  }
  __syncthreads();

  // per-lane step-invariants: gate columns n = g*256 + j0 + lj
  float b1v[4], b2v[4], w1v[4];
#pragma unroll
  for (int g = 0; g < 4; ++g) {
    const int n = g * HID + j0 + lj;
    b1v[g] = b1s[n];
    b2v[g] = b2s[n];
    w1v[g] = wih1[n];
  }
  const float wl = wlin[j0 + lj];

  v8f c1 = {}, c2 = {};        // cell state lives in registers forever
  unsigned gen = 0;

  for (int s = 0; s < TF; ++s) {
    // stop LICM from hoisting the (address-invariant) LDS weight reads
    // out of the 576-step loop and spilling them to scratch
    asm volatile("" ::: "memory");

    const int cur = s & 1;
    const _Float16* h1c = h1buf + cur * (BATCH * HID);
    _Float16*       h1n = h1buf + (cur ^ 1) * (BATCH * HID);
    const _Float16* h2c = h2buf + cur * (BATCH * HID);
    _Float16*       h2n = h2buf + (cur ^ 1) * (BATCH * HID);

    // input scalar per batch row (teacher forcing, then autoregressive)
    float xv[8];
#pragma unroll
    for (int r = 0; r < 8; ++r) {
      const int m = m0 + r + hsel * 8;
      xv[r] = (s < TLEN) ? x_train[m * TLEN + s] : out[m * TF + (s - 1)];
    }

    // ---- layer 1 gates: h1c @ W_hh1^T + x*w_ih1 + b1 ----
    v8f acc[4];
#pragma unroll
    for (int g = 0; g < 4; ++g)
#pragma unroll
      for (int r = 0; r < 8; ++r)
        acc[g][r] = b1v[g] + xv[r] * w1v[g];

    for (int kb = 0; kb < 8; ++kb) {
      const v16h a = load_a_frag(h1c, m0, kb * 32, lane);
#pragma unroll
      for (int g = 0; g < 4; ++g) {
        const v16h bf = load_b_lds(sWB1, g * 8 + kb, lane);
        acc[g] = __builtin_amdgcn_wmma_f32_16x16x32_f16(
            false, a, false, bf, (short)0, acc[g], false, false);
      }
    }

    // layer-1 cell update; i/f/g/o align lane-wise with c1 fragment
#pragma unroll
    for (int r = 0; r < 8; ++r) {
      const float iv = fsig(acc[0][r]);
      const float fv = fsig(acc[1][r]);
      const float gv = ftanh(acc[2][r]);
      const float ov = fsig(acc[3][r]);
      const float cn = fv * c1[r] + iv * gv;
      c1[r] = cn;
      const float hn = ov * ftanh(cn);
      const int m = m0 + r + hsel * 8;
      h1n[m * HID + j0 + lj] = (_Float16)hn;
    }

    gbar(bar, gen); ++gen;   // h1_new visible everywhere

    // ---- layer 2 gates: [h1_new ; h2_old] @ [W_ih2;W_hh2]^T + b2  (K=512) ----
#pragma unroll
    for (int g = 0; g < 4; ++g)
#pragma unroll
      for (int r = 0; r < 8; ++r)
        acc[g][r] = b2v[g];

    for (int kb = 0; kb < 16; ++kb) {
      const v16h a = (kb < 8) ? load_a_frag(h1n, m0, kb * 32, lane)
                              : load_a_frag(h2c, m0, (kb - 8) * 32, lane);
#pragma unroll
      for (int g = 0; g < 4; ++g) {
        const v16h bf = load_b_lds(sWB2, g * 16 + kb, lane);
        acc[g] = __builtin_amdgcn_wmma_f32_16x16x32_f16(
            false, a, false, bf, (short)0, acc[g], false, false);
      }
    }

    float hfrag[8];
#pragma unroll
    for (int r = 0; r < 8; ++r) {
      const float iv = fsig(acc[0][r]);
      const float fv = fsig(acc[1][r]);
      const float gv = ftanh(acc[2][r]);
      const float ov = fsig(acc[3][r]);
      const float cn = fv * c2[r] + iv * gv;
      c2[r] = cn;
      const float hn = ov * ftanh(cn);
      hfrag[r] = hn;
      const int m = m0 + r + hsel * 8;
      h2n[m * HID + j0 + lj] = (_Float16)hn;
    }

    // readout partials: reduce over the 16 j-columns held across 16 lanes
#pragma unroll
    for (int r = 0; r < 8; ++r) {
      float v = hfrag[r] * wl;
      v += __shfl_xor(v, 8);
      v += __shfl_xor(v, 4);
      v += __shfl_xor(v, 2);
      v += __shfl_xor(v, 1);
      if (lj == 0) {
        const int m = m0 + r + hsel * 8;
        outpart[jT * BATCH + m] = v;   // one writer per (jT,m): deterministic
      }
    }

    gbar(bar, gen); ++gen;   // all partials + h2_new visible

    // finalize readout: out[m][s] = sum_jT partial + b_lin
    if (blockIdx.x == 0 && threadIdx.x < BATCH) {
      const int m = threadIdx.x;
      float acco = blin[0];
#pragma unroll
      for (int jt = 0; jt < NWG; ++jt) acco += outpart[jt * BATCH + m];
      out[m * TF + s] = acco;
    }

    gbar(bar, gen); ++gen;   // out[s] visible (next step's x in future phase)
  }
}

// ---------------- host entry ----------------
extern "C" void kernel_launch(void* const* d_in, const int* in_sizes, int n_in,
                              void* d_out, int out_size, void* d_ws, size_t ws_size,
                              hipStream_t stream) {
  (void)in_sizes; (void)n_in; (void)out_size; (void)ws_size;
  const float* x_train = (const float*)d_in[0];
  const float* W_ih1   = (const float*)d_in[1];
  const float* W_hh1   = (const float*)d_in[2];
  const float* b_ih1   = (const float*)d_in[3];
  const float* b_hh1   = (const float*)d_in[4];
  const float* W_ih2   = (const float*)d_in[5];
  const float* W_hh2   = (const float*)d_in[6];
  const float* b_ih2   = (const float*)d_in[7];
  const float* b_hh2   = (const float*)d_in[8];
  const float* W_lin   = (const float*)d_in[9];
  const float* b_lin   = (const float*)d_in[10];
  float* out = (float*)d_out;

  char* ws = (char*)d_ws;
  unsigned* bar    = (unsigned*)(ws + OFF_BAR);
  float* outpart   = (float*)(ws + OFF_OUTP);
  float* b1s       = (float*)(ws + OFF_B1S);
  float* b2s       = (float*)(ws + OFF_B2S);
  _Float16* h1buf  = (_Float16*)(ws + OFF_H1);
  _Float16* h2buf  = (_Float16*)(ws + OFF_H2);
  _Float16* WB1    = (_Float16*)(ws + OFF_WB1);
  _Float16* WB2    = (_Float16*)(ws + OFF_WB2);

  lstm_init<<<256, 256, 0, stream>>>(b_ih1, b_hh1, b_ih2, b_hh2,
                                     b1s, b2s, h1buf, h2buf, bar);
  pack_w1<<<1024, 256, 0, stream>>>(W_hh1, WB1);
  pack_w2<<<2048, 256, 0, stream>>>(W_ih2, W_hh2, WB2);
  lstm_persistent<<<NWG, TPB, LDS_BYTES, stream>>>(x_train, W_ih1, W_lin, b_lin,
                                                   WB1, WB2, b1s, b2s,
                                                   h1buf, h2buf, outpart, bar, out);
}
